// VQVAE_26233660244628
// MI455X (gfx1250) — matrix operand order repaired and checked by reading, not compiled, not run on hardware
//
#include <hip/hip_runtime.h>

// ---------------------------------------------------------------------------
// VQ-VAE forward for MI455X (gfx1250, wave32).
//  * VQ distance GEMM (4.3 GFLOP): v_wmma_f32_16x16x32_f16 fused with argmin,
//    codebook staged in LDS (320KB/WGP) with bank-conflict-free padding.
//  * Decoder conv1 (64->16 3x3, 1.2 GFLOP): implicit-GEMM on WMMA,
//    weights repacked to f16 in LDS once per block.
//  * Remaining convs are <=300 MFLOP each: direct f32 VALU.
// ---------------------------------------------------------------------------

typedef __attribute__((ext_vector_type(16))) _Float16 v16h;
typedef __attribute__((ext_vector_type(8)))  _Float16 v8h;
typedef __attribute__((ext_vector_type(8)))  float    v8f;

__device__ __forceinline__ float leaky(float v) { return v >= 0.0f ? v : 0.01f * v; }

#define WMMA_F16(A, B, C) \
    __builtin_amdgcn_wmma_f32_16x16x32_f16(false, (A), false, (B), (short)0, (C), false, false)

// ---------------------------------------------------------------------------
// Encoder conv1: (64,1,128,128) -> (64,8,64,64), 4x4 stride 2 pad 1
// ---------------------------------------------------------------------------
__global__ __launch_bounds__(256) void enc_conv1(const float* __restrict__ x,
                                                 const float* __restrict__ w,
                                                 const float* __restrict__ b,
                                                 float* __restrict__ out) {
    int idx = blockIdx.x * 256 + threadIdx.x;            // 2097152 outputs
    int px = idx & 63, py = (idx >> 6) & 63, c = (idx >> 12) & 7, n = idx >> 15;
    float s = b[c];
#pragma unroll
    for (int ky = 0; ky < 4; ++ky) {
        int iy = 2 * py - 1 + ky;
        if (iy < 0 || iy >= 128) continue;
#pragma unroll
        for (int kx = 0; kx < 4; ++kx) {
            int ix = 2 * px - 1 + kx;
            if (ix < 0 || ix >= 128) continue;
            s += x[(n * 128 + iy) * 128 + ix] * w[(c * 4 + ky) * 4 + kx];
        }
    }
    out[idx] = leaky(s);
}

// ---------------------------------------------------------------------------
// Encoder conv2: (64,8,64,64) -> (64,16,32,32), 4x4 stride 2 pad 1
// ---------------------------------------------------------------------------
__global__ __launch_bounds__(256) void enc_conv2(const float* __restrict__ in,
                                                 const float* __restrict__ w,
                                                 const float* __restrict__ b,
                                                 float* __restrict__ out) {
    int idx = blockIdx.x * 256 + threadIdx.x;            // 1048576 outputs
    int px = idx & 31, py = (idx >> 5) & 31, c = (idx >> 10) & 15, n = idx >> 14;
    float s = b[c];
#pragma unroll
    for (int ky = 0; ky < 4; ++ky) {
        int iy = 2 * py - 1 + ky;
        if (iy < 0 || iy >= 64) continue;
#pragma unroll
        for (int kx = 0; kx < 4; ++kx) {
            int ix = 2 * px - 1 + kx;
            if (ix < 0 || ix >= 64) continue;
#pragma unroll
            for (int ci = 0; ci < 8; ++ci)
                s += in[((n * 8 + ci) * 64 + iy) * 64 + ix] *
                     w[((c * 8 + ci) * 4 + ky) * 4 + kx];
        }
    }
    out[idx] = leaky(s);
}

// ---------------------------------------------------------------------------
// Encoder conv3: (64,16,32,32) -> (64,16,32,32), 3x3 stride 1 pad 1
// ---------------------------------------------------------------------------
__global__ __launch_bounds__(256) void enc_conv3(const float* __restrict__ in,
                                                 const float* __restrict__ w,
                                                 const float* __restrict__ b,
                                                 float* __restrict__ out) {
    int idx = blockIdx.x * 256 + threadIdx.x;            // 1048576 outputs
    int px = idx & 31, py = (idx >> 5) & 31, c = (idx >> 10) & 15, n = idx >> 14;
    float s = b[c];
#pragma unroll
    for (int ky = 0; ky < 3; ++ky) {
        int iy = py - 1 + ky;
        if (iy < 0 || iy >= 32) continue;
#pragma unroll
        for (int kx = 0; kx < 3; ++kx) {
            int ix = px - 1 + kx;
            if (ix < 0 || ix >= 32) continue;
#pragma unroll
            for (int ci = 0; ci < 16; ++ci)
                s += in[((n * 16 + ci) * 32 + iy) * 32 + ix] *
                     w[((c * 16 + ci) * 3 + ky) * 3 + kx];
        }
    }
    out[idx] = leaky(s);
}

// ---------------------------------------------------------------------------
// Encoder conv4 (1x1, 16->64), writes latents in NHWC: h[m][64], m=((n*32+y)*32+x)
// ---------------------------------------------------------------------------
__global__ __launch_bounds__(256) void enc_conv4_nhwc(const float* __restrict__ in,
                                                      const float* __restrict__ w,
                                                      const float* __restrict__ b,
                                                      float* __restrict__ h) {
    int idx = blockIdx.x * 256 + threadIdx.x;            // 65536*64 outputs
    int co = idx & 63;
    int m  = idx >> 6;
    int px = m & 31, py = (m >> 5) & 31, n = m >> 10;
    float s = b[co];
#pragma unroll
    for (int ci = 0; ci < 16; ++ci)
        s += in[((n * 16 + ci) * 32 + py) * 32 + px] * w[co * 16 + ci];
    h[idx] = leaky(s);
}

// ---------------------------------------------------------------------------
// Codebook prep: f32 [512][64] -> f16 copy + per-entry squared norms
// ---------------------------------------------------------------------------
__global__ __launch_bounds__(256) void codebook_prep(const float* __restrict__ cbf,
                                                     _Float16* __restrict__ cbh,
                                                     float* __restrict__ cnorm) {
    int k = blockIdx.x * 256 + threadIdx.x;              // 512 entries
    if (k >= 512) return;
    float s = 0.0f;
#pragma unroll
    for (int d = 0; d < 64; ++d) {
        float v = cbf[k * 64 + d];
        s += v * v;
        cbh[k * 64 + d] = (_Float16)v;
    }
    cnorm[k] = s;
}

// ---------------------------------------------------------------------------
// Fused VQ: WMMA distance GEMM + in-register argmin.
// One wave per 16-row tile of flat latents [65536][64]; 8 waves / block.
// Codebook staged in LDS in 2 phases of 256 entries (keeps static LDS < 64KB);
// entry stride padded to 72 halves (36 dwords, gcd(36,64)=4) so the 32-lane
// b128 B-fragment read covers all 64 banks exactly twice (optimal).
// dist = ||c||^2 - 2 f.c  (||f||^2 is row-constant, irrelevant for argmin).
// ---------------------------------------------------------------------------
#define CB_STRIDE 72   // halves per entry in LDS (64 data + 8 pad)

__global__ __launch_bounds__(256) void vq_argmin(const float* __restrict__ h,
                                                 const _Float16* __restrict__ cb,
                                                 const float* __restrict__ cnorm,
                                                 int* __restrict__ inds) {
    __shared__ _Float16 scb[256 * CB_STRIDE];            // 36864 B
    __shared__ float    scn[256];

    const int wave = threadIdx.x >> 5;
    const int lane = threadIdx.x & 31;
    const int tile = blockIdx.x * 8 + wave;              // 4096 tiles total
    const int rowBase = tile * 16;

    // --- A fragments (16x32 f16 layout, ISA 7.12.2): lane holds row (lane&15),
    //     halves 0-7 = K kbase..kbase+7, halves 8-15 = K kbase+16..kbase+23.
    const int arow  = rowBase + (lane & 15);
    const int kbase = (lane >> 4) * 8;
    const float* ap = h + arow * 64;
    v16h a0, a1;
#pragma unroll
    for (int g = 0; g < 2; ++g)
#pragma unroll
        for (int j = 0; j < 8; ++j) {
            a0[g * 8 + j] = (_Float16)ap[kbase + 16 * g + j];        // K in [0,32)
            a1[g * 8 + j] = (_Float16)ap[32 + kbase + 16 * g + j];   // K in [32,64)
        }

    // C/D layout: lane owns column (lane&15) and rows r + 8*(lane>>4).
    const int ncol = lane & 15;
    const int kseg = (lane >> 4) * 8;                    // B: K kseg..+7 and kseg+16..+23 per half-wave

    float minv[8];
    int   mini[8];
#pragma unroll
    for (int r = 0; r < 8; ++r) { minv[r] = 3.4e38f; mini[r] = 0; }

    for (int p = 0; p < 2; ++p) {                        // 2 phases x 256 entries
        __syncthreads();                                 // previous phase fully consumed
        // cooperative stage: 256 entries x 128B = 2048 uint4 chunks
        const uint4* src = (const uint4*)cb + p * 2048;
        for (int c = threadIdx.x; c < 2048; c += 256) {
            int e = c >> 3, j = c & 7;
            *(uint4*)((char*)scb + e * (CB_STRIDE * 2) + j * 16) = src[c];
        }
        scn[threadIdx.x & 255] = cnorm[p * 256 + (threadIdx.x & 255)];
        __syncthreads();

#pragma unroll 4
        for (int c = 0; c < 16; ++c) {                   // 256 entries / 16 cols
            const int el    = c * 16 + ncol;             // entry within phase
            const int entry = p * 256 + el;
            const _Float16* bp = scb + el * CB_STRIDE;
            // B fragments (32x16 f16): lane = column, halves = same K order as A.
            v8h l0 = *(const v8h*)(bp + kseg);
            v8h h0 = *(const v8h*)(bp + kseg + 16);      // K in [0,32)
            v8h l1 = *(const v8h*)(bp + 32 + kseg);
            v8h h1 = *(const v8h*)(bp + 32 + kseg + 16); // K in [32,64)
            v16h b0, b1;
#pragma unroll
            for (int i = 0; i < 8; ++i) {
                b0[i] = l0[i]; b0[8 + i] = h0[i];
                b1[i] = l1[i]; b1[8 + i] = h1[i];
            }

            v8f acc = {};
            acc = WMMA_F16(a0, b0, acc);
            acc = WMMA_F16(a1, b1, acc);

            const float cn = scn[el];
#pragma unroll
            for (int r = 0; r < 8; ++r) {
                float dist = cn - 2.0f * acc[r];
                if (dist < minv[r] || (dist == minv[r] && entry < mini[r])) {
                    minv[r] = dist;
                    mini[r] = entry;
                }
            }
        }
    }

    // Reduce across the 16 lanes that share each row set (width-16 xor shuffles).
#pragma unroll
    for (int off = 8; off >= 1; off >>= 1) {
#pragma unroll
        for (int r = 0; r < 8; ++r) {
            float ov = __shfl_xor(minv[r], off, 16);
            int   oi = __shfl_xor(mini[r], off, 16);
            if (ov < minv[r] || (ov == minv[r] && oi < mini[r])) {
                minv[r] = ov;
                mini[r] = oi;
            }
        }
    }
    if ((lane & 15) == 0) {
        const int rs = (lane >> 4) * 8;                  // lane 0 -> rows 0-7, lane 16 -> 8-15
#pragma unroll
        for (int r = 0; r < 8; ++r)
            inds[rowBase + rs + r] = mini[r];
    }
}

// ---------------------------------------------------------------------------
// Gather quantized vectors (straight-through forward value == codebook[ind])
// and accumulate sum((q - lat)^2) for the VQ loss.
// ---------------------------------------------------------------------------
__global__ __launch_bounds__(256) void gather_loss(const float* __restrict__ cbf,
                                                   const int* __restrict__ inds,
                                                   const float* __restrict__ h,
                                                   float* __restrict__ q,
                                                   float* __restrict__ loss_acc) {
    __shared__ float sdata[256];
    int idx = blockIdx.x * 256 + threadIdx.x;            // 65536*64 elements
    int d = idx & 63;
    int m = idx >> 6;
    float qv = cbf[inds[m] * 64 + d];
    q[idx] = qv;
    float diff = qv - h[idx];
    sdata[threadIdx.x] = diff * diff;
    __syncthreads();
#pragma unroll
    for (int s = 128; s > 0; s >>= 1) {
        if (threadIdx.x < s) sdata[threadIdx.x] += sdata[threadIdx.x + s];
        __syncthreads();
    }
    if (threadIdx.x == 0) atomicAdd(loss_acc, sdata[0]);
}

// ---------------------------------------------------------------------------
// Decoder conv1 as implicit GEMM on WMMA.
// Tile: M = 16 pixels (half of one image row), N = 16 out-channels,
// K = 9 taps x 64 in-ch, accumulated as 18 x wmma_f32_16x16x32_f16.
// Weights repacked once per block into LDS f16 [tap][co][ci], co-stride 72.
// The ky bound check is wave-uniform (EXEC stays all-1s around WMMA);
// only x-edge lanes zero-fill their A fragment.
// ---------------------------------------------------------------------------
#define W_STRIDE 72    // halves per (tap,co) row in LDS (64 data + 8 pad)

__global__ __launch_bounds__(256) void dec_conv1_wmma(const float* __restrict__ q,
                                                      const float* __restrict__ w,
                                                      const float* __restrict__ b,
                                                      float* __restrict__ out) {
    __shared__ _Float16 wpk[9 * 16 * W_STRIDE];          // 20736 B
    // repack OIHW (16,64,3,3) -> [tap][co][ci] f16
    for (int i = threadIdx.x; i < 9 * 16 * 64; i += 256) {
        int ci  = i & 63;
        int co  = (i >> 6) & 15;
        int tap = i >> 10;
        wpk[(tap * 16 + co) * W_STRIDE + ci] = (_Float16)w[(co * 64 + ci) * 9 + tap];
    }
    __syncthreads();

    const int wave = threadIdx.x >> 5;
    const int lane = threadIdx.x & 31;
    const int tile = blockIdx.x * 8 + wave;              // 4096 tiles: (n,y,xhalf)
    const int xh = tile & 1;
    const int y  = (tile >> 1) & 31;
    const int n  = tile >> 6;

    const int mrow  = lane & 15;                         // A row = pixel in tile
    const int xx0   = xh * 16 + mrow;
    const int kbase = (lane >> 4) * 8;
    const int kseg  = (lane >> 4) * 8;
    const int co    = lane & 15;

    v8f acc = {};
#pragma unroll
    for (int ky = 0; ky < 3; ++ky) {
        const int yy = y - 1 + ky;
        if (yy < 0 || yy >= 32) continue;                // wave-uniform skip
#pragma unroll
        for (int kx = 0; kx < 3; ++kx) {
            const int xx  = xx0 - 1 + kx;
            const bool okx = (xx >= 0) && (xx < 32);
            const int xxc = okx ? xx : 0;
            const float* ap = q + (((n * 32 + yy) * 32 + xxc) * 64);
            v16h a0, a1;
#pragma unroll
            for (int g = 0; g < 2; ++g)
#pragma unroll
                for (int j = 0; j < 8; ++j) {
                    a0[g * 8 + j] = okx ? (_Float16)ap[kbase + 16 * g + j]      : (_Float16)0.0f;
                    a1[g * 8 + j] = okx ? (_Float16)ap[32 + kbase + 16 * g + j] : (_Float16)0.0f;
                }
            const _Float16* bp = wpk + ((ky * 3 + kx) * 16 + co) * W_STRIDE;
            v8h l0 = *(const v8h*)(bp + kseg);
            v8h h0 = *(const v8h*)(bp + kseg + 16);      // ci in [0,32)
            v8h l1 = *(const v8h*)(bp + 32 + kseg);
            v8h h1 = *(const v8h*)(bp + 32 + kseg + 16); // ci in [32,64)
            v16h b0, b1;
#pragma unroll
            for (int i = 0; i < 8; ++i) {
                b0[i] = l0[i]; b0[8 + i] = h0[i];
                b1[i] = l1[i]; b1[8 + i] = h1[i];
            }
            acc = WMMA_F16(a0, b0, acc);
            acc = WMMA_F16(a1, b1, acc);
        }
    }

    // C layout: lane holds co = lane&15, pixels r + 8*(lane>>4).
    const float bias = b[co];
#pragma unroll
    for (int r = 0; r < 8; ++r) {
        int pix = r + ((lane >> 4) << 3);
        out[((n * 16 + co) * 32 + y) * 32 + xh * 16 + pix] = leaky(acc[r] + bias);
    }
}

// ---------------------------------------------------------------------------
// Decoder convT1: (64,16,32,32) -> (64,8,64,64), 4x4 stride 2 pad 1
// out[y,x] = sum in[(y+1-ky)/2, (x+1-kx)/2] * w[ci,co,ky,kx]  (when integral)
// ---------------------------------------------------------------------------
__global__ __launch_bounds__(256) void dec_convt2(const float* __restrict__ in,
                                                  const float* __restrict__ w,
                                                  const float* __restrict__ b,
                                                  float* __restrict__ out) {
    int idx = blockIdx.x * 256 + threadIdx.x;            // 2097152 outputs
    int px = idx & 63, py = (idx >> 6) & 63, co = (idx >> 12) & 7, n = idx >> 15;
    float s = b[co];
#pragma unroll
    for (int ky = 0; ky < 4; ++ky) {
        int t = py + 1 - ky;
        if (t < 0 || (t & 1)) continue;
        int iy = t >> 1;
        if (iy >= 32) continue;
#pragma unroll
        for (int kx = 0; kx < 4; ++kx) {
            int u = px + 1 - kx;
            if (u < 0 || (u & 1)) continue;
            int ix = u >> 1;
            if (ix >= 32) continue;
#pragma unroll
            for (int ci = 0; ci < 16; ++ci)
                s += in[((n * 16 + ci) * 32 + iy) * 32 + ix] *
                     w[((ci * 8 + co) * 4 + ky) * 4 + kx];       // (in,out,kh,kw)
        }
    }
    out[idx] = leaky(s);
}

// ---------------------------------------------------------------------------
// Decoder convT2 + tanh: (64,8,64,64) -> (64,1,128,128)
// ---------------------------------------------------------------------------
__global__ __launch_bounds__(256) void dec_convt3(const float* __restrict__ in,
                                                  const float* __restrict__ w,
                                                  const float* __restrict__ b,
                                                  float* __restrict__ out) {
    int idx = blockIdx.x * 256 + threadIdx.x;            // 1048576 outputs
    int px = idx & 127, py = (idx >> 7) & 127, n = idx >> 14;
    float s = b[0];
#pragma unroll
    for (int ky = 0; ky < 4; ++ky) {
        int t = py + 1 - ky;
        if (t < 0 || (t & 1)) continue;
        int iy = t >> 1;
        if (iy >= 64) continue;
#pragma unroll
        for (int kx = 0; kx < 4; ++kx) {
            int u = px + 1 - kx;
            if (u < 0 || (u & 1)) continue;
            int ix = u >> 1;
            if (ix >= 64) continue;
#pragma unroll
            for (int ci = 0; ci < 8; ++ci)
                s += in[((n * 8 + ci) * 64 + iy) * 64 + ix] *
                     w[(ci * 4 + ky) * 4 + kx];                  // (in=8,out=1,4,4)
        }
    }
    out[idx] = tanhf(s);
}

// ---------------------------------------------------------------------------
// vq_loss = (1 + BETA) * MSE  (commitment and embedding losses are equal in fwd)
// ---------------------------------------------------------------------------
__global__ void finalize_loss(const float* __restrict__ loss_acc,
                              float* __restrict__ out_scalar) {
    if (threadIdx.x == 0)
        out_scalar[0] = 1.25f * (loss_acc[0] / (65536.0f * 64.0f));
}

// ---------------------------------------------------------------------------
// Workspace layout (bytes); buffers reused once dead. Total ~48.1 MB.
// ---------------------------------------------------------------------------
#define OFF_E1    ((size_t)0)          //  8 MB  (64,8,64,64)   reused as d2
#define OFF_E2    ((size_t)8388608)    //  4 MB  (64,16,32,32)  reused as d1
#define OFF_E3    ((size_t)12582912)   //  4 MB  (64,16,32,32)  reused below after conv4
#define OFF_INDS  ((size_t)12582912)   //  256 KB int[65536]
#define OFF_CNORM ((size_t)12845056)   //  2 KB  float[512]
#define OFF_LOSS  ((size_t)12847104)   //  4 B   float accumulator
#define OFF_H     ((size_t)16777216)   // 16 MB  latents NHWC [65536][64]
#define OFF_Q     ((size_t)33554432)   // 16 MB  quantized NHWC
#define OFF_CB    ((size_t)50331648)   // 64 KB  f16 codebook [512][64]

extern "C" void kernel_launch(void* const* d_in, const int* in_sizes, int n_in,
                              void* d_out, int out_size, void* d_ws, size_t ws_size,
                              hipStream_t stream) {
    const float* x    = (const float*)d_in[0];
    const float* ew1  = (const float*)d_in[1];
    const float* eb1  = (const float*)d_in[2];
    const float* ew2  = (const float*)d_in[3];
    const float* eb2  = (const float*)d_in[4];
    const float* ew3  = (const float*)d_in[5];
    const float* eb3  = (const float*)d_in[6];
    const float* ew4  = (const float*)d_in[7];
    const float* eb4  = (const float*)d_in[8];
    const float* cbf  = (const float*)d_in[9];
    const float* dw1w = (const float*)d_in[10];
    const float* db1  = (const float*)d_in[11];
    const float* dw2w = (const float*)d_in[12];
    const float* db2  = (const float*)d_in[13];
    const float* dw3w = (const float*)d_in[14];
    const float* db3  = (const float*)d_in[15];

    char* ws = (char*)d_ws;
    float*    e1    = (float*)(ws + OFF_E1);
    float*    e2    = (float*)(ws + OFF_E2);
    float*    e3    = (float*)(ws + OFF_E3);
    int*      inds  = (int*)(ws + OFF_INDS);
    float*    cnorm = (float*)(ws + OFF_CNORM);
    float*    lossA = (float*)(ws + OFF_LOSS);
    float*    h     = (float*)(ws + OFF_H);
    float*    q     = (float*)(ws + OFF_Q);
    _Float16* cbh   = (_Float16*)(ws + OFF_CB);
    float*    d1    = e2;   // e2 dead after conv3
    float*    d2    = e1;   // e1 dead after conv2
    float*    out   = (float*)d_out;

    enc_conv1<<<8192, 256, 0, stream>>>(x, ew1, eb1, e1);
    enc_conv2<<<4096, 256, 0, stream>>>(e1, ew2, eb2, e2);
    enc_conv3<<<4096, 256, 0, stream>>>(e2, ew3, eb3, e3);
    enc_conv4_nhwc<<<16384, 256, 0, stream>>>(e3, ew4, eb4, h);

    // e3 region is now dead -> holds inds/cnorm/loss
    hipMemsetAsync(lossA, 0, sizeof(float), stream);
    codebook_prep<<<2, 256, 0, stream>>>(cbf, cbh, cnorm);
    vq_argmin<<<512, 256, 0, stream>>>(h, cbh, cnorm, inds);     // 4096 tiles / 8 waves
    gather_loss<<<16384, 256, 0, stream>>>(cbf, inds, h, q, lossA);

    dec_conv1_wmma<<<512, 256, 0, stream>>>(q, dw1w, db1, d1);   // 4096 tiles / 8 waves
    dec_convt2<<<8192, 256, 0, stream>>>(d1, dw2w, db2, d2);
    dec_convt3<<<4096, 256, 0, stream>>>(d2, dw3w, db3, out);
    finalize_loss<<<1, 32, 0, stream>>>(lossA, out + 1048576);
}
